// CustomAttention_18545668784595
// MI455X (gfx1250) — matrix operand order repaired and checked by reading, not compiled, not run on hardware
//
#include <hip/hip_runtime.h>
#include <hip/hip_bf16.h>
#include <hip/hip_fp16.h>

#define NH  8
#define HD  32
#define NP  4
#define EMB 256

typedef __attribute__((ext_vector_type(16))) _Float16 v16h;
typedef __attribute__((ext_vector_type(2)))  _Float16 h2v;
typedef __attribute__((ext_vector_type(8)))  float    v8f;

// ------------------------------------------------------------------
// Kernel 0: convert Wo (f32) -> f16 for WMMA
// ------------------------------------------------------------------
__global__ void k_cvt_f16(const float* __restrict__ src,
                          _Float16* __restrict__ dst, int n) {
  int i = blockIdx.x * blockDim.x + threadIdx.x;
  if (i < n) dst[i] = (_Float16)src[i];
}

// ------------------------------------------------------------------
// Bilinear sample of 32 contiguous channels, zeros padding,
// align_corners=False semantics (x,y already in pixel space).
// img: [bs, nv, EMB] row-major, channel base = h*HD.
// ------------------------------------------------------------------
__device__ __forceinline__ void bilinear32(const float* __restrict__ img,
                                           int b, int nv, int Hl, int Wl,
                                           int chbase, float x, float y,
                                           float* __restrict__ acc) {
  float x0 = floorf(x), y0 = floorf(y);
  int   x0i = (int)x0,  y0i = (int)y0;
  float wx1 = x - x0, wy1 = y - y0;
  float wx0 = 1.0f - wx1, wy0 = 1.0f - wy1;
  #pragma unroll
  for (int c = 0; c < 4; ++c) {
    int cx = x0i + (c & 1);
    int cy = y0i + (c >> 1);
    float w = ((c & 1) ? wx1 : wx0) * ((c >> 1) ? wy1 : wy0);
    bool valid = (cx >= 0) && (cx < Wl) && (cy >= 0) && (cy < Hl);
    float wv = valid ? w : 0.0f;
    int xi = min(max(cx, 0), Wl - 1);
    int yi = min(max(cy, 0), Hl - 1);
    const float4* p4 = (const float4*)(img +
        ((size_t)b * nv + (size_t)yi * Wl + xi) * EMB + chbase);
    #pragma unroll
    for (int i = 0; i < 8; ++i) {
      float4 v = p4[i];
      acc[4*i+0] += wv * v.x; acc[4*i+1] += wv * v.y;
      acc[4*i+2] += wv * v.z; acc[4*i+3] += wv * v.w;
    }
  }
}

// ------------------------------------------------------------------
// Kernel 1: sampled+projected keys from query 0 only (ref: grids[:, :1]).
// 128 threads: thread t -> b = t>>5, h = (t>>2)&7, p = t&3.
// kproj: [bs, NH, NP, HD] f32
// ------------------------------------------------------------------
__global__ void k_keys(const float* __restrict__ query,
                       const float* __restrict__ key,
                       const float* __restrict__ refp,
                       const float* __restrict__ Wk, const float* __restrict__ bk,
                       const float* __restrict__ Woff, const float* __restrict__ boff,
                       const int* __restrict__ ss,
                       float* __restrict__ kproj,
                       int nq, int nv) {
  int t = threadIdx.x;
  int b = t >> 5, h = (t >> 2) & 7, p = t & 3;
  int Hl = ss[0], Wl = ss[1];

  float qv[HD];
  const float* qptr = query + ((size_t)b * nq) * EMB + h * HD;
  #pragma unroll
  for (int d = 0; d < HD; ++d) qv[d] = qptr[d];

  float ox = boff[2*p], oy = boff[2*p+1];
  #pragma unroll
  for (int d = 0; d < HD; ++d) {
    ox += Woff[(2*p)   * HD + d] * qv[d];
    oy += Woff[(2*p+1) * HD + d] * qv[d];
  }
  float rx = refp[((size_t)b * nq) * 2 + 0];
  float ry = refp[((size_t)b * nq) * 2 + 1];
  // loc = ref + off/(W,H); grid=2*loc-1; pixel x = (gx+1)*W/2 - 0.5 = loc_x*W - 0.5
  float x = (rx + ox / (float)Wl) * (float)Wl - 0.5f;
  float y = (ry + oy / (float)Hl) * (float)Hl - 0.5f;

  float ks[HD];
  #pragma unroll
  for (int d = 0; d < HD; ++d) ks[d] = 0.0f;
  bilinear32(key, b, nv, Hl, Wl, h * HD, x, y, ks);

  float* kout = kproj + (((size_t)b * NH + h) * NP + p) * HD;
  for (int dout = 0; dout < HD; ++dout) {
    float acc = bk[dout];
    #pragma unroll
    for (int d = 0; d < HD; ++d) acc += Wk[dout * HD + d] * ks[d];
    kout[dout] = acc;
  }
}

// ------------------------------------------------------------------
// Kernel 2: per-query sampling, projections, softmax, weighted sum.
// One wave per query; lane = h*4 + p. Output pre_out in f16.
// ------------------------------------------------------------------
__global__ __launch_bounds__(256) void k_attend(
    const float* __restrict__ query, const float* __restrict__ value,
    const float* __restrict__ refp,
    const float* __restrict__ Wq, const float* __restrict__ bq,
    const float* __restrict__ Wv, const float* __restrict__ bv,
    const float* __restrict__ Woff, const float* __restrict__ boff,
    const int* __restrict__ ss,
    const float* __restrict__ kproj,
    _Float16* __restrict__ preout,
    int totalQ, int nq, int nv) {
  __shared__ float sWq[HD * HD];
  __shared__ float sWv[HD * HD];
  __shared__ float sWoff[8 * HD];
  __shared__ float sbq[HD], sbv[HD], sboff[8];

  int tid = threadIdx.x;
  for (int i = tid; i < HD * HD; i += 256) { sWq[i] = Wq[i]; sWv[i] = Wv[i]; }
  for (int i = tid; i < 8 * HD; i += 256) sWoff[i] = Woff[i];
  if (tid < HD) { sbq[tid] = bq[tid]; sbv[tid] = bv[tid]; }
  if (tid < 8)  sboff[tid] = boff[tid];
  __syncthreads();

  int wave = tid >> 5, lane = tid & 31;
  int gq = blockIdx.x * 8 + wave;        // global row index (b*nq + q)
  if (gq >= totalQ) return;              // wave-uniform
  int b = gq / nq;
  int h = lane >> 2, p = lane & 3;
  int Hl = ss[0], Wl = ss[1];

  // ---- load this head's query vector
  float qv[HD];
  const float4* q4 = (const float4*)(query + (size_t)gq * EMB + h * HD);
  #pragma unroll
  for (int i = 0; i < 8; ++i) {
    float4 v = q4[i];
    qv[4*i+0] = v.x; qv[4*i+1] = v.y; qv[4*i+2] = v.z; qv[4*i+3] = v.w;
  }

  // ---- sampling offset for (h, p)
  float ox = sboff[2*p], oy = sboff[2*p+1];
  #pragma unroll
  for (int d = 0; d < HD; ++d) {
    ox += sWoff[(2*p)   * HD + d] * qv[d];
    oy += sWoff[(2*p+1) * HD + d] * qv[d];
  }
  float rx = refp[(size_t)gq * 2 + 0];
  float ry = refp[(size_t)gq * 2 + 1];
  float x = (rx + ox / (float)Wl) * (float)Wl - 0.5f;
  float y = (ry + oy / (float)Hl) * (float)Hl - 0.5f;

  // ---- bilinear sample of value
  float vs[HD];
  #pragma unroll
  for (int d = 0; d < HD; ++d) vs[d] = 0.0f;
  bilinear32(value, b, nv, Hl, Wl, h * HD, x, y, vs);

  // ---- attention logit: (Wq q + bq) . kproj[b,h,p,:]
  const float* kb = kproj + (((size_t)b * NH + h) * NP + p) * HD;
  float logit = 0.0f;
  for (int dout = 0; dout < HD; ++dout) {
    float t = sbq[dout];
    #pragma unroll
    for (int d = 0; d < HD; ++d) t += sWq[dout * HD + d] * qv[d];
    logit += t * kb[dout];
  }

  // ---- softmax over 4 points (lanes 4k..4k+3)
  float mx = logit;
  mx = fmaxf(mx, __shfl_xor(mx, 1));
  mx = fmaxf(mx, __shfl_xor(mx, 2));
  float e = __expf(logit - mx);
  float s = e;
  s += __shfl_xor(s, 1);
  s += __shfl_xor(s, 2);
  float w = e / s;

  // ---- value projection, weight, reduce over points, store f16
  _Float16* orow = preout + (size_t)gq * EMB + h * HD;
  for (int dout = 0; dout < HD; ++dout) {
    float t = sbv[dout];
    #pragma unroll
    for (int d = 0; d < HD; ++d) t += sWv[dout * HD + d] * vs[d];
    t *= w;
    t += __shfl_xor(t, 1);
    t += __shfl_xor(t, 2);
    if (p == 0) orow[dout] = (_Float16)t;
  }
}

// ------------------------------------------------------------------
// Kernel 3: out = preF16[M,256] x WoF16^T + bo + residual  (WMMA)
// One wave per 16x16 output tile; K=256 in 8 steps of 16x16x32 f16 WMMA.
// ------------------------------------------------------------------
__global__ __launch_bounds__(256) void k_outproj(
    const _Float16* __restrict__ A,    // [M, 256] f16 pre-output
    const _Float16* __restrict__ Bw,   // [256, 256] f16 Wo (row = out channel)
    const float* __restrict__ bo,      // [256]
    const float* __restrict__ residual,// [M, 256] f32 (= query)
    float* __restrict__ out,           // [M, 256] f32
    int M) {
  const int lane = threadIdx.x & 31;
  const int wave = threadIdx.x >> 5;
  const int ntN  = EMB / 16;                 // 16 column tiles
  const int ntM  = (M + 15) >> 4;
  int tile = blockIdx.x * 8 + wave;
  if (tile >= ntM * ntN) return;             // wave-uniform
  int tm = tile / ntN, tn = tile % ntN;
  int hi  = lane >> 4;                       // lane half: 0 or 1
  int l16 = lane & 15;
  int row  = tm * 16 + l16; if (row >= M) row = M - 1;  // clamp tail reads
  int ncol = tn * 16 + l16;

  const _Float16* arow = A  + (size_t)row  * EMB;
  const _Float16* brow = Bw + (size_t)ncol * EMB;       // B[k][n] = Wo[n][k]

  v8f acc = {};
  #pragma unroll
  for (int k0 = 0; k0 < EMB; k0 += 32) {
    // A fragment (16-bit A 16x32 layout): element e=2r+s holds
    //   k = k0 + (r/4)*16 + hi*8 + (r%4)*2 + s   -> 8 paired dword loads
    v16h a;
    const _Float16* ap = arow + k0 + hi * 8;
    #pragma unroll
    for (int r = 0; r < 8; ++r) {
      int off = ((r >> 2) << 4) + ((r & 3) << 1);
      h2v pr = *(const h2v*)(ap + off);
      a[2*r]   = pr.x;
      a[2*r+1] = pr.y;
    }
    // B fragment (16-bit B 32x16): lane n = l16, elems 0..7 = K k0+hi*8.., elems 8..15 = K k0+16+hi*8..
    typedef __attribute__((ext_vector_type(8))) _Float16 v8h_;
    v8h_ blo = *(const v8h_*)(brow + k0 + hi * 8);
    v8h_ bhi = *(const v8h_*)(brow + k0 + 16 + hi * 8);
    v16h bfrag = __builtin_shufflevector(blo, bhi, 0,1,2,3,4,5,6,7,8,9,10,11,12,13,14,15);
    acc = __builtin_amdgcn_wmma_f32_16x16x32_f16(
        /*neg_a=*/false, a, /*neg_b=*/false, bfrag,
        /*c_mod=*/(short)0, acc, /*reuse_a=*/false, /*reuse_b=*/false);
  }

  // Epilogue: D layout — VGPR v: M = tm*16 + hi*8 + v, N = ncol
  float bias = bo[ncol];
  #pragma unroll
  for (int v = 0; v < 8; ++v) {
    int m = tm * 16 + hi * 8 + v;
    if (m < M) {
      size_t idx = (size_t)m * EMB + ncol;
      out[idx] = acc[v] + bias + residual[idx];
    }
  }
}

// ------------------------------------------------------------------
extern "C" void kernel_launch(void* const* d_in, const int* in_sizes, int n_in,
                              void* d_out, int out_size, void* d_ws, size_t ws_size,
                              hipStream_t stream) {
  const float* query = (const float*)d_in[0];
  const float* key   = (const float*)d_in[1];
  const float* value = (const float*)d_in[2];
  const float* refp  = (const float*)d_in[3];
  const float* Wq    = (const float*)d_in[4];
  const float* bq    = (const float*)d_in[5];
  const float* Wk    = (const float*)d_in[6];
  const float* bk    = (const float*)d_in[7];
  const float* Wv    = (const float*)d_in[8];
  const float* bv    = (const float*)d_in[9];
  const float* Woff  = (const float*)d_in[10];
  const float* boff  = (const float*)d_in[11];
  const float* Wo    = (const float*)d_in[12];
  const float* bo    = (const float*)d_in[13];
  const int*   ss    = (const int*)d_in[14];

  int totalQ = in_sizes[0] / EMB;   // bs * nq
  int totalV = in_sizes[1] / EMB;   // bs * nv
  const int BS = 4;                 // setup_inputs batch size
  int nq = totalQ / BS;
  int nv = totalV / BS;

  // Workspace layout: kproj (16 KB) | Wo f16 (128 KB) | pre_out f16 (~20 MB)
  char* wsb = (char*)d_ws;
  float*     kproj  = (float*)wsb;
  _Float16*  WoH    = (_Float16*)(wsb + (16 << 10));
  _Float16*  preout = (_Float16*)(wsb + (16 << 10) + (128 << 10));

  k_cvt_f16<<<(EMB * EMB + 255) / 256, 256, 0, stream>>>(Wo, WoH, EMB * EMB);

  k_keys<<<1, BS * 32, 0, stream>>>(query, key, refp, Wk, bk, Woff, boff,
                                    ss, kproj, nq, nv);

  k_attend<<<(totalQ + 7) / 8, 256, 0, stream>>>(
      query, value, refp, Wq, bq, Wv, bv, Woff, boff, ss,
      kproj, preout, totalQ, nq, nv);

  int ntM = (totalQ + 15) / 16;
  int tiles = ntM * (EMB / 16);
  k_outproj<<<(tiles + 7) / 8, 256, 0, stream>>>(
      preout, WoH, bo, query, (float*)d_out, totalQ);
}
